// FPLSTM_35940286333315
// MI455X (gfx1250) — compile-verified, matrix operand order checked by reading, not run on hardware
//
#include <hip/hip_runtime.h>

// ---------------- Types for WMMA (gfx1250, wave32) ----------------
typedef __attribute__((ext_vector_type(8)))  __bf16 v8bf;
typedef __attribute__((ext_vector_type(16))) __bf16 v16bf;
typedef __attribute__((ext_vector_type(8)))  float  v8f;

#define T_STEPS 8192
#define IN_DIM  2048
#define MEM_DIM 2048
#define FIVE_M  (5 * MEM_DIM)
#define NWG_REC 64           // persistent workgroups for the recurrence
#define BLK_REC 256          // 8 waves each
#define SLICE   32           // output elements (j / r) owned per workgroup

// ---------------- helpers ----------------
__device__ __forceinline__ unsigned short f2bf(float f) {
  unsigned u = __float_as_uint(f);
  u += 0x7fffu + ((u >> 16) & 1u);   // round-to-nearest-even
  return (unsigned short)(u >> 16);
}

__device__ __forceinline__ float sigm(float x) {
  return 1.0f / (1.0f + __expf(-x));
}

// Load one 16x32 bf16 fragment (A layout; B^T uses identical addressing since
// Wx is stored [N][K] row-major). Per ISA 7.12.2:
//   lanes 0-15:  row = lane,    elems 0..7 = K k0..k0+7,  elems 8..15 = K k0+16..k0+23
//   lanes 16-31: row = lane-16, elems 0..7 = K k0+8..+15, elems 8..15 = K k0+24..+31
__device__ __forceinline__ v16bf load_frag(const unsigned short* __restrict__ base,
                                           int ld, int row0, int k0, unsigned lane) {
  int r  = row0 + (int)(lane & 15u);
  int kb = k0 + (int)((lane >> 4) << 3);
  const v8bf* p = (const v8bf*)(base + (size_t)r * ld + kb);
  v8bf lo = p[0];          // 16B: K kb..kb+7
  v8bf hi = p[2];          // 16B: K kb+16..kb+23
  return __builtin_shufflevector(lo, hi, 0,1,2,3,4,5,6,7,8,9,10,11,12,13,14,15);
}

__device__ __forceinline__ v8f wmma_bf16(v16bf a, v16bf b, v8f c) {
  return __builtin_amdgcn_wmma_f32_16x16x32_bf16(false, a, false, b,
                                                 (short)0, c, false, false);
}

__device__ __forceinline__ void store_tile(float* __restrict__ out, int ldn,
                                           int m, int n, v8f c, float bias) {
#pragma unroll
  for (int v = 0; v < 8; ++v)
    out[(size_t)(m + v) * ldn + n] = c[v] + bias;
}

// Monotonic grid barrier (NWG_REC WGs co-resident on MI455X's WGP array).
__device__ __forceinline__ void gsync(unsigned* bar, unsigned target) {
  __threadfence();
  __syncthreads();
  if (threadIdx.x == 0) {
    atomicAdd(bar, 1u);
    while (__hip_atomic_load(bar, __ATOMIC_ACQUIRE, __HIP_MEMORY_SCOPE_AGENT) < target)
      __builtin_amdgcn_s_sleep(2);
  }
  __syncthreads();
}

// Wave-per-row dot: weight row streams from (L2-resident) global with a
// global_prefetch of the following row; the shared vector is read from LDS.
// float4 accumulator -> 4 independent FMA chains; loop fully unrolled so all
// 16 b128 weight loads issue before the FMA tail.
__device__ __forceinline__ float row_dot(const float* __restrict__ row,
                                         const float4* __restrict__ vecL,  // LDS
                                         const float* __restrict__ next_row,
                                         unsigned lane) {
  if (next_row) {
    __builtin_prefetch(next_row + lane * 32, 0, 3);          // 128B granules
    __builtin_prefetch(next_row + 1024 + lane * 32, 0, 3);   // cover full 8KB row
  }
  const float4* wr = (const float4*)row;
  float4 a = make_float4(0.f, 0.f, 0.f, 0.f);
#pragma unroll
  for (int it = 0; it < MEM_DIM / 4 / 32; ++it) {
    int j = (int)lane + it * 32;
    float4 w4 = wr[j];
    float4 v4 = vecL[j];
    a.x = fmaf(w4.x, v4.x, a.x);
    a.y = fmaf(w4.y, v4.y, a.y);
    a.z = fmaf(w4.z, v4.z, a.z);
    a.w = fmaf(w4.w, v4.w, a.w);
  }
  float acc = (a.x + a.y) + (a.z + a.w);
#pragma unroll
  for (int off = 16; off; off >>= 1) acc += __shfl_xor(acc, off, 32);
  return acc;
}

// ---------------- kernels ----------------
__global__ void f32_to_bf16_kernel(const float* __restrict__ in,
                                   unsigned short* __restrict__ out, size_t n) {
  size_t i = (size_t)blockIdx.x * blockDim.x + threadIdx.x;
  size_t stride = (size_t)gridDim.x * blockDim.x;
  for (; i < n; i += stride) out[i] = f2bf(in[i]);
}

// xg[m][n] = sum_k A[m][k] * Wx[n][k] + bx[n]
// Block = 256 threads = 8 waves; wave tile 32x32; WG tile 64(M) x 128(N).
__global__ __launch_bounds__(256) void gemm_xg_kernel(
    const unsigned short* __restrict__ Abf,   // (8192 x 2048) bf16 row-major
    const unsigned short* __restrict__ Wxbf,  // (10240 x 2048) bf16 row-major
    const float* __restrict__ bx,             // (10240)
    float* __restrict__ xg)                   // (8192 x 10240) f32
{
  const int K = IN_DIM, N = FIVE_M;
  unsigned lane = threadIdx.x & 31u;
  unsigned wave = threadIdx.x >> 5;                 // 0..7
  int m0 = blockIdx.y * 64  + (int)(wave & 1u) * 32;
  int n0 = blockIdx.x * 128 + (int)(wave >> 1) * 32;

  v8f c00 = {}, c01 = {}, c10 = {}, c11 = {};
  for (int k = 0; k < K; k += 32) {
    v16bf a0 = load_frag(Abf,  K, m0,      k, lane);
    v16bf a1 = load_frag(Abf,  K, m0 + 16, k, lane);
    v16bf b0 = load_frag(Wxbf, K, n0,      k, lane);
    v16bf b1 = load_frag(Wxbf, K, n0 + 16, k, lane);
    c00 = wmma_bf16(a0, b0, c00);
    c01 = wmma_bf16(a0, b1, c01);
    c10 = wmma_bf16(a1, b0, c10);
    c11 = wmma_bf16(a1, b1, c11);
  }
  int nn  = (int)(lane & 15u);
  int mhi = (int)(lane >> 4) * 8;
  store_tile(xg, N, m0 + mhi,      n0 + nn,      c00, bx[n0 + nn]);
  store_tile(xg, N, m0 + mhi,      n0 + 16 + nn, c01, bx[n0 + 16 + nn]);
  store_tile(xg, N, m0 + 16 + mhi, n0 + nn,      c10, bx[n0 + nn]);
  store_tile(xg, N, m0 + 16 + mhi, n0 + 16 + nn, c11, bx[n0 + 16 + nn]);
}

__global__ void init_state_kernel(float* __restrict__ h, unsigned* __restrict__ bar) {
  int i = blockIdx.x * blockDim.x + threadIdx.x;
  if (i < MEM_DIM) h[i] = 0.0f;
  if (i == 0) *bar = 0u;
}

// Persistent sequential scan, 64 WGs x 256 thr (8 waves). WG w owns slice
// j in [32w, 32w+32). Per step:
//   copy h -> LDS; fused gh rows {j, M+j, 2M+j, 3M+j} for own slice (16/wave)
//   -> LDS; gates + m[j] (workgroup-local; c/i/o/f live in LDS)
//   grid barrier (m visible); copy m -> LDS
//   u rows for own slice (4/wave): Wm@m; c,h update (c stays in LDS all steps)
//   grid barrier (h visible)
// Wh+Wm (80MB) stay L2-resident (192MB L2); global exchange is just h+m (16KB).
__global__ __launch_bounds__(BLK_REC) void recurrence_kernel(
    const float* __restrict__ xg,   // (T x 5M)
    const float* __restrict__ Wh,   // (4M x M)
    const float* __restrict__ bh,   // (4M)
    const float* __restrict__ Wm,   // (M x M)
    const float* __restrict__ bm,   // (M)
    float* __restrict__ mG,         // (M) global exchange
    float* __restrict__ h,          // (M) global state (read by all WGs)
    unsigned* __restrict__ bar)
{
  const int M = MEM_DIM;
  __shared__ float4 vecL[MEM_DIM / 4];            // 8KB: h (phase A) / m (phase C)
  __shared__ float  gh_s[4 * SLICE];
  __shared__ float  iS[SLICE], oS[SLICE], fS[SLICE], cS[SLICE];

  const int      jbase = (int)blockIdx.x * SLICE;
  const unsigned td    = threadIdx.x;
  const unsigned lane  = td & 31u;
  const unsigned wv    = td >> 5;     // wave 0..7 within WG

  if (td < SLICE) cS[td] = 0.0f;
  __syncthreads();

  unsigned target = 0;
  for (int t = 0; t < T_STEPS; ++t) {
    const float* x5 = xg + (size_t)t * FIVE_M;

    // ---- stage h into LDS ----
    {
      const float4* hv = (const float4*)h;
#pragma unroll
      for (int j = (int)td; j < M / 4; j += BLK_REC) vecL[j] = hv[j];
    }
    __syncthreads();

    // ---- fused gh (128 rows/WG, 16 per wave) + gates + m ----
#pragma unroll 1
    for (int i = 0; i < 16; ++i) {
      int lidx = (int)wv * 16 + i;                       // 0..127 within WG
      int r    = (lidx >> 5) * M + jbase + (lidx & 31);  // gate*M + j
      const float* next = nullptr;
      if (i + 1 < 16) {
        int l2 = lidx + 1;
        next = Wh + (size_t)((l2 >> 5) * M + jbase + (l2 & 31)) * M;
      }
      float acc = row_dot(Wh + (size_t)r * M, vecL, next, lane);
      if (lane == 0) gh_s[lidx] = acc + bh[r];
    }
    __syncthreads();

    if (td < SLICE) {
      int j = jbase + (int)td;
      float i_ = sigm(x5[j]         + gh_s[td]);
      float o_ = sigm(x5[M + j]     + gh_s[SLICE + td]);
      float z_ = sigm(x5[2 * M + j] + gh_s[2 * SLICE + td]);
      float f_ = sigm(x5[3 * M + j] + gh_s[3 * SLICE + td]);
      iS[td] = i_; oS[td] = o_; fS[td] = f_;
      mG[j] = z_ * tanhf(cS[td]);
    }
    target += NWG_REC; gsync(bar, target);   // m complete everywhere

    // ---- stage m into LDS ----
    {
      const float4* mv = (const float4*)mG;
#pragma unroll
      for (int j = (int)td; j < M / 4; j += BLK_REC) vecL[j] = mv[j];
    }
    __syncthreads();

    // ---- u = tanh(ux + Wm@m + bm); c,h update (32 rows/WG, 4 per wave) ----
#pragma unroll 1
    for (int i = 0; i < 4; ++i) {
      int lr = (int)wv * 4 + i;                // 0..31 within slice
      int r  = jbase + lr;
      const float* next = (i + 1 < 4) ? (Wm + (size_t)(r + 1) * M) : nullptr;
      float acc = row_dot(Wm + (size_t)r * M, vecL, next, lane);
      if (lane == 0) {
        float u  = tanhf(x5[4 * M + r] + acc + bm[r]);
        float cn = iS[lr] * u + fS[lr] * cS[lr];
        cS[lr] = cn;
        h[r] = oS[lr] * tanhf(cn);
      }
    }
    target += NWG_REC; gsync(bar, target);   // h complete everywhere
  }
}

__global__ void copy_out_kernel(const float* __restrict__ h, float* __restrict__ out) {
  int i = blockIdx.x * blockDim.x + threadIdx.x;
  if (i < MEM_DIM) out[i] = h[i];
}

// ---------------- host launch ----------------
extern "C" void kernel_launch(void* const* d_in, const int* in_sizes, int n_in,
                              void* d_out, int out_size, void* d_ws, size_t ws_size,
                              hipStream_t stream) {
  (void)in_sizes; (void)n_in; (void)out_size; (void)ws_size;
  const float* inputs = (const float*)d_in[0];  // (8192, 2048)
  const float* Wx     = (const float*)d_in[1];  // (10240, 2048)
  const float* bx     = (const float*)d_in[2];  // (10240)
  const float* Wh     = (const float*)d_in[3];  // (8192, 2048)
  const float* bh     = (const float*)d_in[4];  // (8192)
  const float* Wm     = (const float*)d_in[5];  // (2048, 2048)
  const float* bm     = (const float*)d_in[6];  // (2048)
  float* out = (float*)d_out;

  // workspace carve-up (all chunks 256B aligned by construction)
  char* ws = (char*)d_ws;
  size_t off = 0;
  float* xg = (float*)(ws + off);          off += (size_t)T_STEPS * FIVE_M * 4;   // 335.5 MB
  unsigned short* Abf  = (unsigned short*)(ws + off); off += (size_t)T_STEPS * IN_DIM * 2;  // 33.5 MB
  unsigned short* Wxbf = (unsigned short*)(ws + off); off += (size_t)FIVE_M * IN_DIM * 2;   // 41.9 MB
  float* mG = (float*)(ws + off);          off += (size_t)MEM_DIM * 4;
  float* hS = (float*)(ws + off);          off += (size_t)MEM_DIM * 4;
  unsigned* bar = (unsigned*)(ws + off);   off += 256;

  // 1) fp32 -> bf16 operand copies for WMMA
  size_t nA = (size_t)T_STEPS * IN_DIM;
  size_t nW = (size_t)FIVE_M * IN_DIM;
  f32_to_bf16_kernel<<<4096, 256, 0, stream>>>(inputs, Abf, nA);
  f32_to_bf16_kernel<<<4096, 256, 0, stream>>>(Wx, Wxbf, nW);

  // 2) xg = inputs @ Wx^T + bx via v_wmma_f32_16x16x32_bf16
  dim3 gg(FIVE_M / 128, T_STEPS / 64);
  gemm_xg_kernel<<<gg, 256, 0, stream>>>(Abf, Wxbf, bx, xg);

  // 3) init global state + barrier (ws is poisoned by harness)
  init_state_kernel<<<(MEM_DIM + 255) / 256, 256, 0, stream>>>(hS, bar);

  // 4) persistent sequential recurrence
  recurrence_kernel<<<NWG_REC, BLK_REC, 0, stream>>>(
      xg, Wh, bh, Wm, bm, mG, hS, bar);

  // 5) emit h_T
  copy_out_kernel<<<(MEM_DIM + 255) / 256, 256, 0, stream>>>(hS, out);
}
